// ResBlock_84275848282324
// MI455X (gfx1250) — compile-verified, matrix operand order checked by reading, not run on hardware
//
#include <hip/hip_runtime.h>
#include <math.h>

// ============================================================================
// ResBlock (binarized TCN residual block) — MI455X / gfx1250
//
// STRUCTURAL COLLAPSE: conv2's input is ste_sign(relu(...)) == +1 everywhere
// (larq sign(0)=+1, relu >= 0). Hence conv2 output is a per-column constant
// depending only on how many causal taps are in-range:
//     phase 0 (t <  4): taps {k=2}          -> S = cs2
//     phase 1 (4<=t<8): taps {k=1,2}        -> S = cs1+cs2
//     phase 2 (t >= 8): taps {k=0,1,2}      -> S = cs0+cs1+cs2
// where cs_k[co] = sum_ci sign(w2[k,ci,co])  (exact integer, |.|<=256).
// conv1 / w1 / bn1 are dead code. Final: out = relu(bn2(S) + x), elementwise.
//
// Roofline: read x (134MB) + write out (134MB) = 268MB @ 23.3 TB/s ~= 11.5us.
// The kernel is pure streaming; the only matrix math left (column sums of
// sign(w2)) is done with v_wmma_i32_16x16x64_iu8: A = all-ones int8 16x64,
// B = sign(w2) as +-1 int8 64x16  =>  D[M,N] = sum_K B[K,N] (replicated over
// M, invariant to the K<->byte permutation inside B's lane layout).
// ============================================================================

#define BN_EPS   1e-3f
#define CCH      256
#define TLEN     2048
#define NBATCH   64

typedef __attribute__((ext_vector_type(8))) int   v8i;
typedef __attribute__((ext_vector_type(4))) float v4f;   // native vector: OK for
                                                         // __builtin_nontemporal_*

// ---------------------------------------------------------------------------
// Kernel A: build bias[3][256] = (S_phase - mean2)*rsqrt(var2+eps) + beta2
// One block, 8 waves (wave32). 48 tasks = {k in 0..2} x {ntile in 0..15};
// each task does 4 chained WMMA_I32_16X16X64_IU8 over the 256-deep ci axis.
// ---------------------------------------------------------------------------
__global__ void __launch_bounds__(256) bias_table_kernel(
    const float* __restrict__ w2,    // [3, 256, 256]
    const float* __restrict__ beta2, // [256]
    const float* __restrict__ mean2, // [256]
    const float* __restrict__ var2,  // [256]
    float* __restrict__ bias)        // out: [3, 256] in d_ws
{
    __shared__ float cs[3][CCH];

    const int tid  = threadIdx.x;
    const int wave = tid >> 5;   // warpSize == 32 on gfx1250
    const int lane = tid & 31;
    const int half = lane >> 4;  // lane group 0..1 (covers two 32-wide K halves)
    const int ncol = lane & 15;  // B/D column N = lane & 15

    // All 8 waves run the task loop uniformly -> EXEC all-ones at every WMMA.
    for (int task = wave; task < 48; task += 8) {
        const int k     = task >> 4;   // kernel tap 0..2
        const int ntile = task & 15;   // 16-column tile of C_out
        const int co    = ntile * 16 + ncol;

        v8i a;                          // A = 16x64 all-ones int8
#pragma unroll
        for (int i = 0; i < 8; ++i) a[i] = 0x01010101;

        v8i acc = {};
#pragma unroll
        for (int chunk = 0; chunk < 4; ++chunk) {   // ci in chunks of 64
            const int cibase = chunk * 64 + half * 32;
            v8i b;
#pragma unroll
            for (int v = 0; v < 8; ++v) {
                unsigned word = 0;
#pragma unroll
                for (int j = 0; j < 4; ++j) {
                    const int ci = cibase + v * 4 + j;
                    const float w = w2[(k * CCH + ci) * CCH + co];
                    const unsigned byte = (w >= 0.0f) ? 0x01u : 0xFFu; // +1 / -1
                    word |= byte << (8 * j);
                }
                b[v] = (int)word;
            }
            // (sgn_a, A, sgn_b, B, C, reuse_a, reuse_b)
            acc = __builtin_amdgcn_wmma_i32_16x16x64_iu8(
                true, a, true, b, acc, false, false);
        }
        // D layout: VGPR0 holds (M=0, N=lane) for lanes 0-15 and (M=8, N=lane-16)
        // for lanes 16-31. A is all-ones so every row is identical -> both lane
        // halves hold the same column sum; duplicate same-value LDS store is benign.
        cs[k][co] = (float)acc[0];
    }
    __syncthreads();

    // Fuse bn2 into per-phase bias constants. Exact same formula as reference.
    {
        const int co = tid;
        const float c0 = cs[0][co], c1 = cs[1][co], c2 = cs[2][co];
        const float inv = 1.0f / sqrtf(var2[co] + BN_EPS);
        const float m = mean2[co], be = beta2[co];
        bias[0 * CCH + co] = (c2 - m) * inv + be;              // t < 4
        bias[1 * CCH + co] = ((c1 + c2) - m) * inv + be;       // 4 <= t < 8
        bias[2 * CCH + co] = ((c0 + c1 + c2) - m) * inv + be;  // t >= 8
    }
}

// ---------------------------------------------------------------------------
// Kernel B: out[b,t,c] = relu(x[b,t,c] + bias[phase(t)][c])
// Pure HBM streaming: v4f (b128) loads, LDS bias table (aligned 16 ->
// ds_load_b128), global_prefetch_b8 ahead of the stream, non-temporal stores
// (write-once stream; don't pollute L2).
// ---------------------------------------------------------------------------
__global__ void __launch_bounds__(256) resblock_apply_kernel(
    const v4f* __restrict__ x4,
    const float* __restrict__ bias,
    v4f* __restrict__ o4,
    int total4)
{
    __shared__ __attribute__((aligned(16))) float lb[3 * CCH];
    for (int i = threadIdx.x; i < 3 * CCH; i += 256) lb[i] = bias[i];
    __syncthreads();

    const int stride = blockDim.x * gridDim.x;
    for (int i = blockIdx.x * blockDim.x + threadIdx.x; i < total4; i += stride) {
        // Prefetch next grid-stride iteration (speculative; OOB is dropped).
        __builtin_prefetch(&x4[i + stride], 0, 0);

        const int c4    = i & 63;                 // v4f index within C=256
        const int t     = (i >> 6) & (TLEN - 1);  // time index
        const int phase = (t >= 8) ? 2 : ((t >= 4) ? 1 : 0);

        const v4f xv = x4[i];
        const v4f bv = *(const v4f*)&lb[phase * CCH + c4 * 4];

        v4f r;
        r.x = fmaxf(xv.x + bv.x, 0.0f);
        r.y = fmaxf(xv.y + bv.y, 0.0f);
        r.z = fmaxf(xv.z + bv.z, 0.0f);
        r.w = fmaxf(xv.w + bv.w, 0.0f);
        __builtin_nontemporal_store(r, &o4[i]);
    }
}

// ---------------------------------------------------------------------------
// Host entry. Input order (setup_inputs): x, w1, beta1, mean1, var1,
//                                         w2, beta2, mean2, var2
// w1/bn1 are provably dead (see header comment).
// ---------------------------------------------------------------------------
extern "C" void kernel_launch(void* const* d_in, const int* in_sizes, int n_in,
                              void* d_out, int out_size, void* d_ws, size_t ws_size,
                              hipStream_t stream)
{
    const float* x     = (const float*)d_in[0];
    const float* w2    = (const float*)d_in[5];
    const float* beta2 = (const float*)d_in[6];
    const float* mean2 = (const float*)d_in[7];
    const float* var2  = (const float*)d_in[8];
    float* bias = (float*)d_ws;     // 3*256 floats, written fully each call
    float* out  = (float*)d_out;

    bias_table_kernel<<<1, 256, 0, stream>>>(w2, beta2, mean2, var2, bias);

    const int total4 = (NBATCH * TLEN * CCH) / 4;   // 8,388,608 v4f
    const int blocks = 2048;                        // 16 v4f per thread
    resblock_apply_kernel<<<blocks, 256, 0, stream>>>(
        (const v4f*)x, bias, (v4f*)out, total4);
}
